// Simpl_55722905698630
// MI455X (gfx1250) — matrix operand order repaired and checked by reading, not compiled, not run on hardware
//
#include <hip/hip_runtime.h>
#include <math.h>

#define NN 50000
#define NE 500000
#define DD 128
#define DFF 512
#define NH 8

typedef __attribute__((ext_vector_type(16))) _Float16 v16h;
typedef __attribute__((ext_vector_type(8)))  float    v8f;

// ---------------- WMMA fragment helpers (CDNA5 wave32 layouts) ----------------

// A fragment: 16x32 f16 tile at column k0 of a row-major f16 matrix in LDS.
// lane<16: row M=lane, K base 0 ; lane>=16: K base 8 ; VGPR4-7 add +16.
// Per lane this is two contiguous 16B runs -> compiler emits 2x ds_load_b128.
__device__ __forceinline__ v16h a_frag_lds(const _Float16* A, int lda, int k0, int lane) {
  const int m  = lane & 15;
  const int kb = (lane >> 4) << 3;           // 0 or 8
  const _Float16* row = A + m * lda + k0;
  v16h a;
#pragma unroll
  for (int j = 0; j < 8; ++j) {
    const int kk = ((j >> 2) << 4) + kb + ((j & 3) << 1);
    a[2 * j]     = row[kk];
    a[2 * j + 1] = row[kk + 1];
  }
  return a;
}

// B fragment from pre-swizzled weight: one contiguous 32B load per lane.
// Swizzled layout: frag(kt, nt) -> 32 lanes x 16 f16 contiguous.
__device__ __forceinline__ v16h b_frag_sw(const _Float16* Bsw, int ntiles,
                                          int kt, int nt, int lane) {
  const _Float16* p = Bsw + ((size_t)((kt * ntiles + nt) * 32 + lane) << 4);
  return *(const v16h*)p;
}

__device__ __forceinline__ void zero_acc(v8f acc[8]) {
#pragma unroll
  for (int i = 0; i < 8; ++i)
#pragma unroll
    for (int j = 0; j < 8; ++j) acc[i][j] = 0.f;
}

// 16 x 128 output tile: A (16 x kTot f16, LDS) x swizzled B, n-tiles nt0..nt0+7
__device__ __forceinline__ void gemm8(const _Float16* Alds, int lda,
                                      const _Float16* Bsw, int ntiles, int nt0,
                                      int kTot, v8f acc[8], int lane) {
  const int ksteps = kTot >> 5;
  for (int kt = 0; kt < ksteps; ++kt) {
    const v16h a = a_frag_lds(Alds, lda, kt << 5, lane);
#pragma unroll
    for (int j = 0; j < 8; ++j) {
      const v16h b = b_frag_sw(Bsw, ntiles, kt, nt0 + j, lane);
      acc[j] = __builtin_amdgcn_wmma_f32_16x16x32_f16(
          false, a, false, b, (short)0, acc[j], false, false);
    }
  }
}

// C 16x16 f32: col N = lane&15, rows mh..mh+7 with mh = (lane>=16)?8:0
__device__ __forceinline__ void store_c_lds(float* C, const v8f acc[8], int lane) {
  const int n  = lane & 15;
  const int mh = (lane >> 4) << 3;
#pragma unroll
  for (int nt = 0; nt < 8; ++nt)
#pragma unroll
    for (int j = 0; j < 8; ++j)
      C[(mh + j) * DD + nt * 16 + n] = acc[nt][j];
}

// order-preserving float<->uint encoding for atomicMax-based segment max
__device__ __forceinline__ unsigned enc_f(float f) {
  unsigned u = __float_as_uint(f);
  return (u & 0x80000000u) ? ~u : (u | 0x80000000u);
}
__device__ __forceinline__ float dec_f(unsigned u) {
  unsigned v = (u & 0x80000000u) ? (u & 0x7FFFFFFFu) : ~u;
  return __uint_as_float(v);
}

// ---------------- weight swizzle: f32 [K x N] -> f16 WMMA-B fragment layout ---

__global__ void k_swz(const float* __restrict__ W, _Float16* __restrict__ Bsw,
                      int K, int Ncols) {
  int i = blockIdx.x * 256 + threadIdx.x;
  if (i >= K * Ncols) return;
  const int e      = i & 15;          // element within lane fragment
  const int lane   = (i >> 4) & 31;
  const int tile   = i >> 9;          // (kt * ntiles + nt)
  const int ntiles = Ncols >> 4;
  const int kt = tile / ntiles;
  const int nt = tile - kt * ntiles;
  const int n  = nt * 16 + (lane & 15);
  const int k  = kt * 32 + ((e >> 3) << 4) + ((lane >> 4) << 3) + (e & 7);
  Bsw[i] = (_Float16)W[(size_t)k * Ncols + n];
}

__global__ void k_zero(unsigned* __restrict__ p, int n) {
  int i = blockIdx.x * 256 + threadIdx.x;
  if (i < n) p[i] = 0u;
}

// ---------------- node q-projection: xq = x @ q_w ----------------

__global__ void __launch_bounds__(64)
k_xq(const float* __restrict__ x, const _Float16* __restrict__ Wq,
     float* __restrict__ xq) {
  __shared__ __attribute__((aligned(16))) _Float16 Ax[2][16 * DD];
  const int lane = threadIdx.x & 31;
  const int w    = threadIdx.x >> 5;
  const int tile = blockIdx.x * 2 + w;
  const bool active = tile < (NN / 16);
  const int i0 = tile * 16;
  if (active) {
    for (int idx = lane; idx < 16 * DD; idx += 32)
      Ax[w][idx] = (_Float16)x[(size_t)i0 * DD + idx];
  }
  __syncthreads();
  if (active) {
    v8f acc[8]; zero_acc(acc);
    gemm8(Ax[w], DD, Wq, 8, 0, DD, acc, lane);
    const int n = lane & 15, mh = (lane >> 4) << 3;
#pragma unroll
    for (int nt = 0; nt < 8; ++nt)
#pragma unroll
      for (int j = 0; j < 8; ++j)
        xq[(size_t)(i0 + mh + j) * DD + nt * 16 + n] = acc[nt][j];
  }
}

// ---------------- fused per-edge kernel: mem / edge-update / k·q logits / v ----

__global__ void __launch_bounds__(64)
k_edge(const float* __restrict__ x, const int* __restrict__ ei,
       const float* __restrict__ edge_attr,
       const float* __restrict__ mp_b, const float* __restrict__ mp_g, const float* __restrict__ mp_be,
       const float* __restrict__ eu_b, const float* __restrict__ eu_g, const float* __restrict__ eu_be,
       const float* __restrict__ en_g, const float* __restrict__ en_be,
       const _Float16* __restrict__ Wmp, const _Float16* __restrict__ Weu,
       const _Float16* __restrict__ Wk,  const _Float16* __restrict__ Wv,
       const float* __restrict__ xq,
       float* __restrict__ logits, unsigned* __restrict__ mxEnc,
       _Float16* __restrict__ vbuf, float* __restrict__ edge_out) {
  __shared__ __attribute__((aligned(16))) _Float16 catT[2][16 * 384];
  __shared__ __attribute__((aligned(16))) _Float16 memT[2][16 * DD];
  __shared__ __attribute__((aligned(16))) float    scr [2][16 * DD];
  __shared__ int srcT[2][16], dstT[2][16];

  const int lane = threadIdx.x & 31;
  const int w    = threadIdx.x >> 5;
  const int tile = blockIdx.x * 2 + w;      // exact: NE/16 = 31250 = 2*15625
  const int e0   = tile * 16;

  if (lane < 16) {
    srcT[w][lane] = ei[e0 + lane];
    dstT[w][lane] = ei[NE + e0 + lane];
  }
  __syncthreads();

  // stage concat [x[dst] | x[src] | edge_attr] as f16 (16 x 384)
  for (int r = 0; r < 16; ++r) {
    const float* xd = x + (size_t)dstT[w][r] * DD;
    const float* xs = x + (size_t)srcT[w][r] * DD;
    const float* ea = edge_attr + (size_t)(e0 + r) * DD;
    int c = lane;
#pragma unroll
    for (int i = 0; i < 12; ++i, c += 32) {
      float val = (c < 128) ? xd[c] : (c < 256) ? xs[c - 128] : ea[c - 256];
      catT[w][r * 384 + c] = (_Float16)val;
    }
  }
  __syncthreads();

  v8f acc[8];

  // mem = relu(LN(cat @ mp_w + mp_b))
  zero_acc(acc);
  gemm8(catT[w], 384, Wmp, 8, 0, 384, acc, lane);
  store_c_lds(scr[w], acc, lane);
  __syncthreads();
  if (lane < 16) {
    float* crow = scr[w] + lane * DD;
    float s = 0.f;
    for (int i = 0; i < DD; ++i) s += crow[i] + mp_b[i];
    const float mean = s * (1.f / DD);
    float vv = 0.f;
    for (int i = 0; i < DD; ++i) { float d = crow[i] + mp_b[i] - mean; vv += d * d; }
    const float rinv = rsqrtf(vv * (1.f / DD) + 1e-5f);
    for (int i = 0; i < DD; ++i) {
      float t = (crow[i] + mp_b[i] - mean) * rinv * mp_g[i] + mp_be[i];
      memT[w][lane * DD + i] = (_Float16)fmaxf(t, 0.f);
    }
  }
  __syncthreads();

  // delta = relu(LN(mem @ eu_w + eu_b)); edge_out = LN(edge_attr + delta)
  zero_acc(acc);
  gemm8(memT[w], DD, Weu, 8, 0, DD, acc, lane);
  store_c_lds(scr[w], acc, lane);
  __syncthreads();
  if (lane < 16) {
    const int e = e0 + lane;
    float* crow = scr[w] + lane * DD;
    float s = 0.f;
    for (int i = 0; i < DD; ++i) s += crow[i] + eu_b[i];
    const float mean = s * (1.f / DD);
    float vv = 0.f;
    for (int i = 0; i < DD; ++i) { float d = crow[i] + eu_b[i] - mean; vv += d * d; }
    const float rinv = rsqrtf(vv * (1.f / DD) + 1e-5f);
    const float* ea = edge_attr + (size_t)e * DD;
    float s2 = 0.f;
    for (int i = 0; i < DD; ++i) {
      float t = (crow[i] + eu_b[i] - mean) * rinv * eu_g[i] + eu_be[i];
      float o = ea[i] + fmaxf(t, 0.f);
      crow[i] = o;
      s2 += o;
    }
    const float mean2 = s2 * (1.f / DD);
    float vv2 = 0.f;
    for (int i = 0; i < DD; ++i) { float d = crow[i] - mean2; vv2 += d * d; }
    const float rinv2 = rsqrtf(vv2 * (1.f / DD) + 1e-5f);
    for (int i = 0; i < DD; ++i)
      edge_out[(size_t)e * DD + i] = (crow[i] - mean2) * rinv2 * en_g[i] + en_be[i];
  }
  __syncthreads();

  // k = mem @ k_w ; per-head logits + scatter-max over dst
  zero_acc(acc);
  gemm8(memT[w], DD, Wk, 8, 0, DD, acc, lane);
  store_c_lds(scr[w], acc, lane);
  __syncthreads();
  if (lane < 16) {
    const int e = e0 + lane;
    const int d = dstT[w][lane];
    const float* qrow = xq + (size_t)d * DD;
    const float* krow = scr[w] + lane * DD;
#pragma unroll
    for (int h = 0; h < NH; ++h) {
      float s = 0.f;
#pragma unroll
      for (int j = 0; j < 16; ++j) s += qrow[h * 16 + j] * krow[h * 16 + j];
      s *= 0.25f;   // 1/sqrt(16)
      logits[(size_t)e * NH + h] = s;
      atomicMax(&mxEnc[d * NH + h], enc_f(s));
    }
  }
  __syncthreads();

  // v = mem @ v_w -> f16 buffer
  zero_acc(acc);
  gemm8(memT[w], DD, Wv, 8, 0, DD, acc, lane);
  store_c_lds(scr[w], acc, lane);
  __syncthreads();
  for (int idx = lane; idx < 16 * DD; idx += 32)
    vbuf[(size_t)e0 * DD + idx] = (_Float16)scr[w][idx];
}

// ---------------- softmax pieces ----------------

__global__ void k_mxfin(unsigned* __restrict__ mxEnc, int n) {
  int i = blockIdx.x * 256 + threadIdx.x;
  if (i < n) {
    float f = dec_f(mxEnc[i]);
    if (!isfinite(f)) f = 0.f;
    ((float*)mxEnc)[i] = f;   // reuse buffer as float max
  }
}

__global__ void k_expsum(float* logits, const int* __restrict__ dst,
                         const float* __restrict__ mxF, float* __restrict__ denom) {
  int i = blockIdx.x * 256 + threadIdx.x;
  if (i < NE * NH) {
    int e = i >> 3, h = i & 7;
    int d = dst[e];
    float ev = __expf(logits[i] - mxF[d * NH + h]);
    logits[i] = ev;           // in place: logits -> exp values
    atomicAdd(&denom[d * NH + h], ev);
  }
}

// ---------------- msg = (attn * v) @ o_w, scatter-add into agg ----------------

__global__ void __launch_bounds__(64)
k_msg(const int* __restrict__ ei, const float* __restrict__ evals,
      const float* __restrict__ denom, const _Float16* __restrict__ vbuf,
      const _Float16* __restrict__ Wo, float* __restrict__ agg) {
  __shared__ __attribute__((aligned(16))) _Float16 vA[2][16 * DD];
  __shared__ float attnT[2][16 * NH];
  __shared__ int dstT[2][16];
  const int lane = threadIdx.x & 31;
  const int w    = threadIdx.x >> 5;
  const int tile = blockIdx.x * 2 + w;
  const int e0   = tile * 16;

  if (lane < 16) {
    const int e = e0 + lane;
    const int d = ei[NE + e];
    dstT[w][lane] = d;
#pragma unroll
    for (int h = 0; h < NH; ++h)
      attnT[w][lane * NH + h] =
          evals[(size_t)e * NH + h] / (denom[d * NH + h] + 1e-16f);
  }
  __syncthreads();
  for (int idx = lane; idx < 16 * DD; idx += 32) {
    const int r = idx >> 7, c = idx & 127;
    vA[w][idx] = (_Float16)((float)vbuf[(size_t)e0 * DD + idx] *
                            attnT[w][r * NH + (c >> 4)]);
  }
  __syncthreads();
  v8f acc[8]; zero_acc(acc);
  gemm8(vA[w], DD, Wo, 8, 0, DD, acc, lane);
  const int n = lane & 15, mh = (lane >> 4) << 3;
#pragma unroll
  for (int nt = 0; nt < 8; ++nt)
#pragma unroll
    for (int j = 0; j < 8; ++j)
      atomicAdd(&agg[(size_t)dstT[w][mh + j] * DD + nt * 16 + n], acc[nt][j]);
}

// ---------------- node-side: LN + FFN + LN ----------------

__global__ void __launch_bounds__(64)
k_node(const float* __restrict__ x, const float* __restrict__ agg,
       const float* __restrict__ n1_g, const float* __restrict__ n1_be,
       const float* __restrict__ f1_b, const float* __restrict__ f2_b,
       const float* __restrict__ n2_g, const float* __restrict__ n2_be,
       const _Float16* __restrict__ Wf1, const _Float16* __restrict__ Wf2,
       float* __restrict__ out_h) {
  __shared__ __attribute__((aligned(16))) _Float16 h16[2][16 * DD];
  __shared__ __attribute__((aligned(16))) float    hf [2][16 * DD];
  __shared__ __attribute__((aligned(16))) _Float16 aT [2][16 * DFF];
  const int lane = threadIdx.x & 31;
  const int w    = threadIdx.x >> 5;
  const int tile = blockIdx.x * 2 + w;
  const bool active = tile < (NN / 16);
  const int i0 = tile * 16;

  if (active && lane < 16) {
    const size_t base = (size_t)(i0 + lane) * DD;
    float s = 0.f;
    for (int c = 0; c < DD; ++c) s += x[base + c] + agg[base + c];
    const float mean = s * (1.f / DD);
    float vv = 0.f;
    for (int c = 0; c < DD; ++c) { float d = x[base + c] + agg[base + c] - mean; vv += d * d; }
    const float rinv = rsqrtf(vv * (1.f / DD) + 1e-5f);
    for (int c = 0; c < DD; ++c) {
      float t = (x[base + c] + agg[base + c] - mean) * rinv * n1_g[c] + n1_be[c];
      hf[w][lane * DD + c]  = t;
      h16[w][lane * DD + c] = (_Float16)t;
    }
  }
  __syncthreads();

  if (active) {
    const int n = lane & 15, mh = (lane >> 4) << 3;
#pragma unroll
    for (int g = 0; g < 4; ++g) {           // 512 output cols in 4 chunks of 128
      v8f acc[8]; zero_acc(acc);
      gemm8(h16[w], DD, Wf1, 32, g * 8, DD, acc, lane);
#pragma unroll
      for (int nt = 0; nt < 8; ++nt)
#pragma unroll
        for (int j = 0; j < 8; ++j) {
          const int col = g * DD + nt * 16 + n;
          aT[w][(mh + j) * DFF + col] = (_Float16)fmaxf(acc[nt][j] + f1_b[col], 0.f);
        }
    }
  }
  __syncthreads();

  float* scr2 = (float*)aT[w];              // reuse after GEMM2 (deps via acc)
  if (active) {
    v8f acc[8]; zero_acc(acc);
    gemm8(aT[w], DFF, Wf2, 8, 0, DFF, acc, lane);
    store_c_lds(scr2, acc, lane);
  }
  __syncthreads();
  if (active && lane < 16) {
    float* crow = scr2 + lane * DD;
    float s = 0.f;
    for (int c = 0; c < DD; ++c) {
      float t = hf[w][lane * DD + c] + crow[c] + f2_b[c];
      crow[c] = t; s += t;
    }
    const float mean = s * (1.f / DD);
    float vv = 0.f;
    for (int c = 0; c < DD; ++c) { float d = crow[c] - mean; vv += d * d; }
    const float rinv = rsqrtf(vv * (1.f / DD) + 1e-5f);
    for (int c = 0; c < DD; ++c)
      out_h[(size_t)(i0 + lane) * DD + c] = (crow[c] - mean) * rinv * n2_g[c] + n2_be[c];
  }
}

// ---------------- host ----------------

extern "C" void kernel_launch(void* const* d_in, const int* in_sizes, int n_in,
                              void* d_out, int out_size, void* d_ws, size_t ws_size,
                              hipStream_t stream) {
  (void)in_sizes; (void)n_in; (void)out_size; (void)ws_size;

  const float* x         = (const float*)d_in[0];
  const int*   ei        = (const int*)  d_in[1];
  const float* edge_attr = (const float*)d_in[2];
  const float* mp_w = (const float*)d_in[3];
  const float* mp_b = (const float*)d_in[4];
  const float* mp_g = (const float*)d_in[5];
  const float* mp_be= (const float*)d_in[6];
  const float* q_w  = (const float*)d_in[7];
  const float* k_w  = (const float*)d_in[8];
  const float* v_w  = (const float*)d_in[9];
  const float* o_w  = (const float*)d_in[10];
  const float* eu_w = (const float*)d_in[11];
  const float* eu_b = (const float*)d_in[12];
  const float* eu_g = (const float*)d_in[13];
  const float* eu_be= (const float*)d_in[14];
  const float* en_g = (const float*)d_in[15];
  const float* en_be= (const float*)d_in[16];
  const float* f1_w = (const float*)d_in[17];
  const float* f1_b = (const float*)d_in[18];
  const float* f2_w = (const float*)d_in[19];
  const float* f2_b = (const float*)d_in[20];
  const float* n1_g = (const float*)d_in[21];
  const float* n1_be= (const float*)d_in[22];
  const float* n2_g = (const float*)d_in[23];
  const float* n2_be= (const float*)d_in[24];

  float* out_h = (float*)d_out;                       // [N, D]
  float* out_e = (float*)d_out + (size_t)NN * DD;     // [E, D]

  // workspace layout
  char* ws = (char*)d_ws;
  size_t off = 0;
  auto alloc = [&](size_t bytes) -> void* {
    size_t a = (off + 255) & ~(size_t)255;
    off = a + bytes;
    return (void*)(ws + a);
  };
  _Float16* Wmp = (_Float16*)alloc((size_t)384 * DD * 2);
  _Float16* Weu = (_Float16*)alloc((size_t)DD * DD * 2);
  _Float16* Wq  = (_Float16*)alloc((size_t)DD * DD * 2);
  _Float16* Wk  = (_Float16*)alloc((size_t)DD * DD * 2);
  _Float16* Wv  = (_Float16*)alloc((size_t)DD * DD * 2);
  _Float16* Wo  = (_Float16*)alloc((size_t)DD * DD * 2);
  _Float16* Wf1 = (_Float16*)alloc((size_t)DD * DFF * 2);
  _Float16* Wf2 = (_Float16*)alloc((size_t)DFF * DD * 2);
  float*    xq    = (float*)alloc((size_t)NN * DD * 4);
  float*    logit = (float*)alloc((size_t)NE * NH * 4);
  unsigned* mxE   = (unsigned*)alloc((size_t)NN * NH * 4);
  float*    denom = (float*)alloc((size_t)NN * NH * 4);
  float*    agg   = (float*)alloc((size_t)NN * DD * 4);
  _Float16* vbuf  = (_Float16*)alloc((size_t)NE * DD * 2);

  // swizzle weights f32 -> f16 WMMA-B fragment layout
  auto swz = [&](const float* s, _Float16* d, int K, int Ncols) {
    int n = K * Ncols;
    k_swz<<<(n + 255) / 256, 256, 0, stream>>>(s, d, K, Ncols);
  };
  swz(mp_w, Wmp, 384, DD);
  swz(eu_w, Weu, DD, DD);
  swz(q_w,  Wq,  DD, DD);
  swz(k_w,  Wk,  DD, DD);
  swz(v_w,  Wv,  DD, DD);
  swz(o_w,  Wo,  DD, DD);
  swz(f1_w, Wf1, DD, DFF);
  swz(f2_w, Wf2, DFF, DD);

  // zero accumulators (mxE==0 encodes "less than every float")
  k_zero<<<(NN * NH + 255) / 256, 256, 0, stream>>>(mxE, NN * NH);
  k_zero<<<(NN * NH + 255) / 256, 256, 0, stream>>>((unsigned*)denom, NN * NH);
  k_zero<<<(NN * DD + 255) / 256, 256, 0, stream>>>((unsigned*)agg, NN * DD);

  const int nodeTiles = NN / 16;   // 3125
  const int edgeTiles = NE / 16;   // 31250

  k_xq<<<(nodeTiles + 1) / 2, 64, 0, stream>>>(x, Wq, xq);

  k_edge<<<edgeTiles / 2, 64, 0, stream>>>(
      x, ei, edge_attr, mp_b, mp_g, mp_be, eu_b, eu_g, eu_be, en_g, en_be,
      Wmp, Weu, Wk, Wv, xq, logit, mxE, vbuf, out_e);

  k_mxfin<<<(NN * NH + 255) / 256, 256, 0, stream>>>(mxE, NN * NH);

  k_expsum<<<(NE * NH + 255) / 256, 256, 0, stream>>>(
      logit, ei + NE, (const float*)mxE, denom);

  k_msg<<<edgeTiles / 2, 64, 0, stream>>>(ei, logit, denom, vbuf, Wo, agg);

  k_node<<<(nodeTiles + 1) / 2, 64, 0, stream>>>(
      x, agg, n1_g, n1_be, f1_b, f2_b, n2_g, n2_be, Wf1, Wf2, out_h);
}
